// LSTMCoordinator_66666482368895
// MI455X (gfx1250) — matrix operand order repaired and checked by reading, not compiled, not run on hardware
//
#include <hip/hip_runtime.h>
#include <hip/hip_bf16.h>
#include <stdint.h>

// Problem constants (from reference)
#define T_STEPS 200
#define BATCH   16384
#define IN_DIM  2
#define HID     20
#define ACTS    3
#define GPAD    128   // 4 gates (i,f,g,o), each padded 20 -> 32 rows
#define KPAD    32    // K padded: 20 hidden (+2 x-slots for layer0) -> 32
#define WAVES   8     // waves per block (wave32)
#define TPB     256

typedef _Float16 v16h __attribute__((ext_vector_type(16)));
typedef _Float16 v8h  __attribute__((ext_vector_type(8)));
typedef _Float16 v4h  __attribute__((ext_vector_type(4)));
typedef float    v8f  __attribute__((ext_vector_type(8)));

#define LOG2E 1.4426950408889634f
#define LN2   0.6931471805599453f

static __device__ __forceinline__ v8f wmma_f16(v16h a, v16h b, v8f c) {
  // D = A(16x32 f16) x B(32x16 f16) + C(16x16 f32)
  return __builtin_amdgcn_wmma_f32_16x16x32_f16(false, a, false, b, (short)0, c,
                                                false, false);
}

static __device__ __forceinline__ void wait_lds() {
  asm volatile("s_wait_dscnt 0" ::: "memory");
}

// Branchless hardware-native activations (v_exp_f32 / v_rcp_f32, no EXEC
// divergence between WMMAs; XDL WMMA co-executes with these TRANS/VALU ops).
static __device__ __forceinline__ float fast_sigmoid(float x) {
  return __builtin_amdgcn_rcpf(1.0f + __builtin_amdgcn_exp2f(-LOG2E * x));
}
static __device__ __forceinline__ float fast_tanh(float x) {
  // tanh(x) = 1 - 2/(exp(2x)+1); saturates correctly at +/-1 for |x| large
  return 1.0f -
         2.0f * __builtin_amdgcn_rcpf(1.0f +
                                      __builtin_amdgcn_exp2f(2.0f * LOG2E * x));
}
static __device__ __forceinline__ float fast_log(float x) {
  return __builtin_amdgcn_logf(x) * LN2;
}
static __device__ __forceinline__ float fast_exp(float x) {
  return __builtin_amdgcn_exp2f(x * LOG2E);
}

// A-fragment (f16 16x32): lanes 0-15 -> M=lane, K = 0..15;
// lanes 16-31 -> M=lane-16, K = 16..31. LDS tile is [rows][KPAD] f16.
static __device__ __forceinline__ v16h load_a(const _Float16* W, int m, int ln,
                                              int koff) {
  const _Float16* row = W + (m * 16 + ln) * KPAD + 2 * koff;
  v16h a;
  _Float16* pa = (_Float16*)&a;
  *(v16h*)(pa)    = *(const v16h*)(row);
  return a;
}

// C/D bias fragment: lane holds N=ln column, rows M = m*16 + off + r.
static __device__ __forceinline__ v8f load_bias(const float* bsrc, int m, int off) {
  v8f c;
  float* pc = (float*)&c;
  *(float4*)(pc)     = *(const float4*)(bsrc + m * 16 + off);
  *(float4*)(pc + 4) = *(const float4*)(bsrc + m * 16 + off + 4);
  return c;
}

struct U2 { uint32_t x, y; };
static __device__ __forceinline__ uint32_t rotl32(uint32_t v, int n) {
  return (v << n) | (v >> (32 - n));
}
// Threefry-2x32-20 (JAX PRNG core) for jax.random.key(42) reproduction.
static __device__ U2 threefry2x32(uint32_t k0, uint32_t k1, uint32_t c0,
                                  uint32_t c1) {
  uint32_t ks[3] = {k0, k1, k0 ^ k1 ^ 0x1BD11BDAu};
  uint32_t x0 = c0 + k0, x1 = c1 + k1;
  const int RA[4] = {13, 15, 26, 6}, RB[4] = {17, 29, 16, 24};
#pragma unroll
  for (int g = 0; g < 5; ++g) {
    const int* R = (g & 1) ? RB : RA;
#pragma unroll
    for (int i = 0; i < 4; ++i) { x0 += x1; x1 = rotl32(x1, R[i]); x1 ^= x0; }
    x0 += ks[(g + 1) % 3];
    x1 += ks[(g + 2) % 3] + (uint32_t)(g + 1);
  }
  return {x0, x1};
}

__global__ void __launch_bounds__(TPB)
lstm_wmma_kernel(const float* __restrict__ x,
                 const float* __restrict__ h0_in, const float* __restrict__ c0_in,
                 const float* __restrict__ Wih0, const float* __restrict__ Whh0,
                 const float* __restrict__ bih0, const float* __restrict__ bhh0,
                 const float* __restrict__ Wih1, const float* __restrict__ Whh1,
                 const float* __restrict__ bih1, const float* __restrict__ bhh1,
                 const float* __restrict__ Wdec, const float* __restrict__ bdec,
                 float* __restrict__ out) {
  // LDS: padded f16 weight matrices in A-layout-friendly row-major [GPAD][KPAD]
  __shared__ __align__(64) _Float16 sW0[GPAD][KPAD];   // [W_hh_0 | W_ih_0 | 0]
  __shared__ __align__(64) _Float16 sWi1[GPAD][KPAD];  // [W_ih_1 | 0]
  __shared__ __align__(64) _Float16 sWh1[GPAD][KPAD];  // [W_hh_1 | 0]
  __shared__ __align__(64) float sB0[GPAD];
  __shared__ __align__(64) float sB1[GPAD];
  // Per-wave h staging tiles: [batch row 0..15][Kpad 0..31] f16 (B-operand layout)
  __shared__ __align__(64) _Float16 sH[WAVES][2][16][KPAD];

  const int tid = threadIdx.x;

  // ---- cooperative weight prep (padded gate rows: p = gate*32 + k) ----
  for (int idx = tid; idx < GPAD * KPAD; idx += TPB) {
    const int p = idx >> 5, c = idx & 31;
    const int g = p >> 5, k = p & 31;
    float w0 = 0.f, wi1 = 0.f, wh1 = 0.f;
    if (k < HID) {
      const int grow = g * HID + k;
      if (c < HID) {
        w0  = Whh0[grow * HID + c];
        wi1 = Wih1[grow * HID + c];
        wh1 = Whh1[grow * HID + c];
      } else if (c < HID + IN_DIM) {
        w0 = Wih0[grow * IN_DIM + (c - HID)];  // x folded into layer0 K-dim
      }
    }
    sW0[p][c]  = (_Float16)w0;
    sWi1[p][c] = (_Float16)wi1;
    sWh1[p][c] = (_Float16)wh1;
  }
  for (int p = tid; p < GPAD; p += TPB) {
    const int g = p >> 5, k = p & 31;
    float b0v = 0.f, b1v = 0.f;
    if (k < HID) {
      const int grow = g * HID + k;
      b0v = bih0[grow] + bhh0[grow];
      b1v = bih1[grow] + bhh1[grow];
    }
    sB0[p] = b0v;
    sB1[p] = b1v;
  }
  for (int idx = tid; idx < WAVES * 2 * 16 * KPAD; idx += TPB)
    ((_Float16*)sH)[idx] = (_Float16)0.f;
  __syncthreads();

  // ---- wave mapping: one wave owns 16 batch columns ----
  const int wave = tid >> 5, lane = tid & 31;
  const int ln = lane & 15, hi = lane >> 4, off = hi * 8;
  const int b = (blockIdx.x * WAVES + wave) * 16 + ln;  // this lane's batch col

  // cell state in registers: even slots k = off + r; odd slots k = 16 + r (hi==0)
  float c0e[8], c1e[8], c0o[4], c1o[4];
#pragma unroll
  for (int r = 0; r < 8; ++r) {
    c0e[r] = c0_in[0 * BATCH * HID + b * HID + off + r];
    c1e[r] = c0_in[1 * BATCH * HID + b * HID + off + r];
    sH[wave][0][ln][off + r] = (_Float16)h0_in[0 * BATCH * HID + b * HID + off + r];
    sH[wave][1][ln][off + r] = (_Float16)h0_in[1 * BATCH * HID + b * HID + off + r];
  }
#pragma unroll
  for (int r = 0; r < 4; ++r) {
    c0o[r] = 0.f; c1o[r] = 0.f;
    if (hi == 0) {
      c0o[r] = c0_in[0 * BATCH * HID + b * HID + 16 + r];
      c1o[r] = c0_in[1 * BATCH * HID + b * HID + 16 + r];
      sH[wave][0][ln][16 + r] = (_Float16)h0_in[0 * BATCH * HID + b * HID + 16 + r];
      sH[wave][1][ln][16 + r] = (_Float16)h0_in[1 * BATCH * HID + b * HID + 16 + r];
    }
  }

  float acc0 = 0.f, acc1 = 0.f, acc2 = 0.f;  // fused decoder partials

  for (int t = 0; t < T_STEPS; ++t) {
    // stage x_t into layer0 K-slots 20/21 (one writer per row)
    if (hi == 0) {
      const float* xp = x + ((size_t)t * BATCH + b) * IN_DIM;
      sH[wave][0][ln][HID]     = (_Float16)xp[0];
      sH[wave][0][ln][HID + 1] = (_Float16)xp[1];
      if (t + 4 < T_STEPS)
        __builtin_prefetch(x + ((size_t)(t + 4) * BATCH + b) * IN_DIM, 0, 3);
    }
    wait_lds();
    // B-operands: lane ln holds column ln, K half selected by lane half
    v16h bh0 = *(const v16h*)(&sH[wave][0][ln][hi * 16]);
    v16h bh1 = *(const v16h*)(&sH[wave][1][ln][hi * 16]);

    // ================= layer 0 =================
    // even tiles m=0,2,4,6 -> this lane's rows are k = off + r for gate m/2
    v8f di = wmma_f16(load_a(&sW0[0][0], 0, ln, off), bh0, load_bias(sB0, 0, off));
    v8f df = wmma_f16(load_a(&sW0[0][0], 2, ln, off), bh0, load_bias(sB0, 2, off));
    v8f dg = wmma_f16(load_a(&sW0[0][0], 4, ln, off), bh0, load_bias(sB0, 4, off));
    v8f dq = wmma_f16(load_a(&sW0[0][0], 6, ln, off), bh0, load_bias(sB0, 6, off));
    v8h hp8;
#pragma unroll
    for (int r = 0; r < 8; ++r) {
      const float iv = fast_sigmoid(di[r]), fv = fast_sigmoid(df[r]);
      const float gv = fast_tanh(dg[r]),    ov = fast_sigmoid(dq[r]);
      c0e[r] = fv * c0e[r] + iv * gv;
      hp8[r] = (_Float16)(ov * fast_tanh(c0e[r]));
    }
    *(v8h*)(&sH[wave][0][ln][off]) = hp8;

    // odd tiles m=1,3,5,7 -> rows k = 16 + off + r (valid hi==0, r<4)
    v8f oi = wmma_f16(load_a(&sW0[0][0], 1, ln, off), bh0, load_bias(sB0, 1, off));
    v8f of = wmma_f16(load_a(&sW0[0][0], 3, ln, off), bh0, load_bias(sB0, 3, off));
    v8f og = wmma_f16(load_a(&sW0[0][0], 5, ln, off), bh0, load_bias(sB0, 5, off));
    v8f oq = wmma_f16(load_a(&sW0[0][0], 7, ln, off), bh0, load_bias(sB0, 7, off));
    v4h hp4;
#pragma unroll
    for (int r = 0; r < 4; ++r) {
      const float iv = fast_sigmoid(oi[r]), fv = fast_sigmoid(of[r]);
      const float gv = fast_tanh(og[r]),    ov = fast_sigmoid(oq[r]);
      c0o[r] = fv * c0o[r] + iv * gv;
      hp4[r] = (_Float16)(ov * fast_tanh(c0o[r]));
    }
    if (hi == 0) *(v4h*)(&sH[wave][0][ln][16]) = hp4;
    wait_lds();
    v16h bh0n = *(const v16h*)(&sH[wave][0][ln][hi * 16]);  // new h0 (+x slots,
                                                            // zeroed in Wih1)
    // ================= layer 1 =================
    v8f ei = wmma_f16(load_a(&sWi1[0][0], 0, ln, off), bh0n, load_bias(sB1, 0, off));
    ei     = wmma_f16(load_a(&sWh1[0][0], 0, ln, off), bh1, ei);
    v8f ef = wmma_f16(load_a(&sWi1[0][0], 2, ln, off), bh0n, load_bias(sB1, 2, off));
    ef     = wmma_f16(load_a(&sWh1[0][0], 2, ln, off), bh1, ef);
    v8f eg = wmma_f16(load_a(&sWi1[0][0], 4, ln, off), bh0n, load_bias(sB1, 4, off));
    eg     = wmma_f16(load_a(&sWh1[0][0], 4, ln, off), bh1, eg);
    v8f eq = wmma_f16(load_a(&sWi1[0][0], 6, ln, off), bh0n, load_bias(sB1, 6, off));
    eq     = wmma_f16(load_a(&sWh1[0][0], 6, ln, off), bh1, eq);
    float h1e[8];
#pragma unroll
    for (int r = 0; r < 8; ++r) {
      const float iv = fast_sigmoid(ei[r]), fv = fast_sigmoid(ef[r]);
      const float gv = fast_tanh(eg[r]),    ov = fast_sigmoid(eq[r]);
      c1e[r] = fv * c1e[r] + iv * gv;
      h1e[r] = ov * fast_tanh(c1e[r]);
      hp8[r] = (_Float16)h1e[r];
    }
    *(v8h*)(&sH[wave][1][ln][off]) = hp8;

    v8f pi = wmma_f16(load_a(&sWi1[0][0], 1, ln, off), bh0n, load_bias(sB1, 1, off));
    pi     = wmma_f16(load_a(&sWh1[0][0], 1, ln, off), bh1, pi);
    v8f pf = wmma_f16(load_a(&sWi1[0][0], 3, ln, off), bh0n, load_bias(sB1, 3, off));
    pf     = wmma_f16(load_a(&sWh1[0][0], 3, ln, off), bh1, pf);
    v8f pg = wmma_f16(load_a(&sWi1[0][0], 5, ln, off), bh0n, load_bias(sB1, 5, off));
    pg     = wmma_f16(load_a(&sWh1[0][0], 5, ln, off), bh1, pg);
    v8f pq = wmma_f16(load_a(&sWi1[0][0], 7, ln, off), bh0n, load_bias(sB1, 7, off));
    pq     = wmma_f16(load_a(&sWh1[0][0], 7, ln, off), bh1, pq);
    float h1o[4];
#pragma unroll
    for (int r = 0; r < 4; ++r) {
      const float iv = fast_sigmoid(pi[r]), fv = fast_sigmoid(pf[r]);
      const float gv = fast_tanh(pg[r]),    ov = fast_sigmoid(pq[r]);
      c1o[r] = fv * c1o[r] + iv * gv;
      h1o[r] = ov * fast_tanh(c1o[r]);
      hp4[r] = (_Float16)h1o[r];
    }
    if (hi == 0) *(v4h*)(&sH[wave][1][ln][16]) = hp4;

    // ---- fused decoder: logit[b,a] += sum_k h1[k] * Wdec[a, t*HID + k] ----
    const int kb = t * HID;
    float pa[3];
#pragma unroll
    for (int a = 0; a < ACTS; ++a) {
      const float* wr = Wdec + a * (T_STEPS * HID) + kb;
      const float4 w0 = *(const float4*)(wr + off);
      const float4 w1 = *(const float4*)(wr + off + 4);
      pa[a] = h1e[0] * w0.x + h1e[1] * w0.y + h1e[2] * w0.z + h1e[3] * w0.w +
              h1e[4] * w1.x + h1e[5] * w1.y + h1e[6] * w1.z + h1e[7] * w1.w;
      if (hi == 0) {
        const float4 w2 = *(const float4*)(wr + 16);
        pa[a] += h1o[0] * w2.x + h1o[1] * w2.y + h1o[2] * w2.z + h1o[3] * w2.w;
      }
    }
    acc0 += pa[0]; acc1 += pa[1]; acc2 += pa[2];
  }
  wait_lds();

  // ---- outputs: action[B], log_prob[B], h_n[2,B,H], c_n[2,B,H] ----
  const size_t oA = 0, oLP = BATCH, oH = 2 * (size_t)BATCH,
               oC = 2 * (size_t)BATCH + 2 * (size_t)BATCH * HID;
#pragma unroll
  for (int r = 0; r < 8; ++r) {
    out[oH + 0 * BATCH * HID + b * HID + off + r] = (float)sH[wave][0][ln][off + r];
    out[oH + 1 * BATCH * HID + b * HID + off + r] = (float)sH[wave][1][ln][off + r];
    out[oC + 0 * BATCH * HID + b * HID + off + r] = c0e[r];
    out[oC + 1 * BATCH * HID + b * HID + off + r] = c1e[r];
  }
  if (hi == 0) {
#pragma unroll
    for (int r = 0; r < 4; ++r) {
      out[oH + b * HID + 16 + r]               = (float)sH[wave][0][ln][16 + r];
      out[oH + BATCH * HID + b * HID + 16 + r] = (float)sH[wave][1][ln][16 + r];
      out[oC + b * HID + 16 + r]               = c0o[r];
      out[oC + BATCH * HID + b * HID + 16 + r] = c1o[r];
    }
  }

  // combine lane halves (same batch column, disjoint k sets)
  const float l0 = acc0 + __shfl_xor(acc0, 16, 32);
  const float l1 = acc1 + __shfl_xor(acc1, 16, 32);
  const float l2 = acc2 + __shfl_xor(acc2, 16, 32);
  if (hi == 0) {
    float lg[3] = {l0 + bdec[0], l1 + bdec[1], l2 + bdec[2]};
    const float m = fmaxf(lg[0], fmaxf(lg[1], lg[2]));
    const float lse = m + fast_log(fast_exp(lg[0] - m) + fast_exp(lg[1] - m) +
                                   fast_exp(lg[2] - m));
    // gumbel-argmax with threefry2x32 key (0,42) == jax.random.key(42)
    const uint32_t n = (uint32_t)BATCH * ACTS, halfn = n >> 1;
    int act = 0;
    float best = -3.4e38f;
#pragma unroll
    for (int a = 0; a < ACTS; ++a) {
      const uint32_t idx = (uint32_t)b * ACTS + (uint32_t)a;
      const uint32_t j = (idx < halfn) ? idx : (idx - halfn);
      const U2 rr = threefry2x32(0u, 42u, j, halfn + j);
      const uint32_t bits = (idx < halfn) ? rr.x : rr.y;
      float u = fmaxf((float)(bits >> 9) * (1.0f / 8388608.0f), 1.1754944e-38f);
      const float sc = lg[a] + (-fast_log(-fast_log(u)));
      if (sc > best) { best = sc; act = a; }
    }
    out[oA + b] = (float)act;
    out[oLP + b] = lg[act] - lse;
  }
}

extern "C" void kernel_launch(void* const* d_in, const int* in_sizes, int n_in,
                              void* d_out, int out_size, void* d_ws, size_t ws_size,
                              hipStream_t stream) {
  (void)in_sizes; (void)n_in; (void)out_size; (void)d_ws; (void)ws_size;
  const float* x    = (const float*)d_in[0];
  const float* h0   = (const float*)d_in[1];
  const float* c0   = (const float*)d_in[2];
  const float* Wih0 = (const float*)d_in[3];
  const float* Whh0 = (const float*)d_in[4];
  const float* bih0 = (const float*)d_in[5];
  const float* bhh0 = (const float*)d_in[6];
  const float* Wih1 = (const float*)d_in[7];
  const float* Whh1 = (const float*)d_in[8];
  const float* bih1 = (const float*)d_in[9];
  const float* bhh1 = (const float*)d_in[10];
  const float* Wdec = (const float*)d_in[11];
  const float* bdec = (const float*)d_in[12];
  float* out = (float*)d_out;

  dim3 grid(BATCH / (WAVES * 16));  // 128 blocks x 8 waves x 16 batch = 16384
  dim3 block(TPB);
  hipLaunchKernelGGL(lstm_wmma_kernel, grid, block, 0, stream,
                     x, h0, c0, Wih0, Whh0, bih0, bhh0,
                     Wih1, Whh1, bih1, bhh1, Wdec, bdec, out);
}
